// L2ContrastiveLoss_41377714930042
// MI455X (gfx1250) — matrix operand order repaired and checked by reading, not compiled, not run on hardware
//
#include <hip/hip_runtime.h>

typedef __attribute__((ext_vector_type(16))) __bf16 v16bf;
typedef __attribute__((ext_vector_type(8)))  float  v8f;

union FragBF { v16bf v; unsigned int w[8]; };

__device__ __forceinline__ unsigned short f32_to_bf16(float f) {
  unsigned int x = __float_as_uint(f);
  x += 0x7fffu + ((x >> 16) & 1u);          // round-to-nearest-even
  return (unsigned short)(x >> 16);
}
__device__ __forceinline__ float bf16f(unsigned short h) {
  return __uint_as_float(((unsigned int)h) << 16);
}

#define TILE    64
#define KC      32
#define PITCH_H 36   // ushort pitch: keeps K-pair u32 reads 4B-aligned, spreads banks

// A-fragment 16x32 bf16: lane<16 -> K base 0, lane>=16 -> +8; v<4: K=2v, v>=4: K=2v+8
__device__ __forceinline__ void load_frag_A(const unsigned short* p, int colA, int kgrp,
                                            FragBF& f) {
#pragma unroll
  for (int v = 0; v < 8; ++v) {
    const int kk = ((v < 4) ? (2 * v) : (2 * v + 8)) + (kgrp << 3);
    f.w[v] = *(const unsigned int*)&p[colA * PITCH_H + kk];
  }
}
// B-fragment 32x16 bf16: lane<16 -> K base 0, lane>=16 -> +8; v<4: K=2v, v>=4: K=2v+8
__device__ __forceinline__ void load_frag_B(const unsigned short* p, int colB, int kgrp,
                                            FragBF& f) {
#pragma unroll
  for (int v = 0; v < 8; ++v) {
    const int kk = ((v < 4) ? (2 * v) : (2 * v + 8)) + (kgrp << 3);
    f.w[v] = *(const unsigned int*)&p[colB * PITCH_H + kk];
  }
}

// 3-term compensated accumulation: (ah+al)(bh+bl) ~ ah*bh + ah*bl + al*bh.
// B sub-tiles processed one at a time to keep the live fragment set small.
__device__ __forceinline__ void mma_pass(const unsigned short* hiA, const unsigned short* loA,
                                         const unsigned short* hiB, const unsigned short* loB,
                                         int wave, int l16, int kgrp, v8f acc[4]) {
  FragBF ah, al;
  load_frag_A(hiA, (wave << 4) + l16, kgrp, ah);
  load_frag_A(loA, (wave << 4) + l16, kgrp, al);
#pragma unroll
  for (int nb = 0; nb < 4; ++nb) {
    FragBF bh, bl;
    load_frag_B(hiB, (nb << 4) + l16, kgrp, bh);
    load_frag_B(loB, (nb << 4) + l16, kgrp, bl);
    acc[nb] = __builtin_amdgcn_wmma_f32_16x16x32_bf16(false, ah.v, false, bh.v,
                                                      (short)0, acc[nb], false, false);
    acc[nb] = __builtin_amdgcn_wmma_f32_16x16x32_bf16(false, ah.v, false, bl.v,
                                                      (short)0, acc[nb], false, false);
    acc[nb] = __builtin_amdgcn_wmma_f32_16x16x32_bf16(false, al.v, false, bh.v,
                                                      (short)0, acc[nb], false, false);
  }
}

// One block computes the (d0,e0) 64x64 tile of BOTH Grams Gx = X^T X and
// Gy = Y^T Y (K loop over N samples), then emits the partial Frobenius dot
// sum(Gx_tile * Gy_tile).  Sum over all 144 blocks == sum_{i,j} <x_i,y_j>^2.
__global__ __launch_bounds__(128)
void gram_frob_kernel(const float* __restrict__ X, const float* __restrict__ Y,
                      float* __restrict__ partialF, int N, int D) {
  __shared__ unsigned short hiXA[TILE * PITCH_H], loXA[TILE * PITCH_H];
  __shared__ unsigned short hiXB[TILE * PITCH_H], loXB[TILE * PITCH_H];
  __shared__ unsigned short hiYA[TILE * PITCH_H], loYA[TILE * PITCH_H];
  __shared__ unsigned short hiYB[TILE * PITCH_H], loYB[TILE * PITCH_H];
  __shared__ float wsum[4];

  const int tid  = threadIdx.x;
  const int wave = tid >> 5;
  const int lane = tid & 31;
  const int l16  = lane & 15;
  const int kgrp = lane >> 4;
  const int d0 = blockIdx.y * TILE;   // Gram row block
  const int e0 = blockIdx.x * TILE;   // Gram col block

  v8f accX[4], accY[4];
  const v8f vzero = {0.f, 0.f, 0.f, 0.f, 0.f, 0.f, 0.f, 0.f};
#pragma unroll
  for (int i = 0; i < 4; ++i) { accX[i] = vzero; accY[i] = vzero; }

  for (int k0 = 0; k0 < N; k0 += KC) {
    __syncthreads();
    // Stage KC x TILE panels (coalesced in d) and do the hi/lo bf16 split
    // ONCE per element here; inner MMA loop then has zero conversion VALU.
#pragma unroll 4
    for (int it = 0; it < 16; ++it) {
      const int e = tid + (it << 7);
      const int k = e >> 6;
      const int d = e & 63;
      const size_t gk = (size_t)(k0 + k) * (size_t)D;
      const int off = d * PITCH_H + k;   // [d][k] so K-pairs are contiguous

      float f; unsigned short h;
      f = X[gk + d0 + d]; h = f32_to_bf16(f);
      hiXA[off] = h; loXA[off] = f32_to_bf16(f - bf16f(h));
      f = X[gk + e0 + d]; h = f32_to_bf16(f);
      hiXB[off] = h; loXB[off] = f32_to_bf16(f - bf16f(h));
      f = Y[gk + d0 + d]; h = f32_to_bf16(f);
      hiYA[off] = h; loYA[off] = f32_to_bf16(f - bf16f(h));
      f = Y[gk + e0 + d]; h = f32_to_bf16(f);
      hiYB[off] = h; loYB[off] = f32_to_bf16(f - bf16f(h));
    }
    __syncthreads();
    mma_pass(hiXA, loXA, hiXB, loXB, wave, l16, kgrp, accX);
    mma_pass(hiYA, loYA, hiYB, loYB, wave, l16, kgrp, accY);
  }

  // Elementwise product of the two accumulator tiles (identical lane layout),
  // deterministic tree reduction to one partial per block.
  float s = 0.f;
#pragma unroll
  for (int nb = 0; nb < 4; ++nb)
#pragma unroll
    for (int v = 0; v < 8; ++v) s += accX[nb][v] * accY[nb][v];
#pragma unroll
  for (int off = 16; off > 0; off >>= 1) s += __shfl_xor(s, off, 32);
  if (lane == 0) wsum[wave] = s;
  __syncthreads();
  if (tid == 0)
    partialF[blockIdx.y * gridDim.x + blockIdx.x] = (wsum[0] + wsum[1]) + (wsum[2] + wsum[3]);
}

// Per-row dots d_i = <x_i, y_i>; emit per-block partials of sum(d_i), sum(d_i^2).
__global__ __launch_bounds__(256)
void rowdot_kernel(const float* __restrict__ X, const float* __restrict__ Y,
                   float* __restrict__ pd1, float* __restrict__ pd2, int N, int D) {
  __shared__ float s1[8], s2[8];
  const int tid  = threadIdx.x;
  const int wave = tid >> 5;
  const int lane = tid & 31;
  const int gw = blockIdx.x * 8 + wave;
  const int nw = gridDim.x * 8;
  float a1 = 0.f, a2 = 0.f;
  for (int row = gw; row < N; row += nw) {
    const float* xr = X + (size_t)row * (size_t)D;
    const float* yr = Y + (size_t)row * (size_t)D;
    float s = 0.f;
    for (int d = lane; d < D; d += 32) s += xr[d] * yr[d];
#pragma unroll
    for (int off = 16; off > 0; off >>= 1) s += __shfl_xor(s, off, 32);
    a1 += s;
    a2 += s * s;
  }
  if (lane == 0) { s1[wave] = a1; s2[wave] = a2; }
  __syncthreads();
  if (tid == 0) {
    float t1 = 0.f, t2 = 0.f;
#pragma unroll
    for (int i = 0; i < 8; ++i) { t1 += s1[i]; t2 += s2[i]; }
    pd1[blockIdx.x] = t1;
    pd2[blockIdx.x] = t2;
  }
}

__global__ void finalize_kernel(const float* __restrict__ pF, int nF,
                                const float* __restrict__ pd1,
                                const float* __restrict__ pd2, int nd,
                                float* __restrict__ out, int N) {
  if (threadIdx.x == 0 && blockIdx.x == 0) {
    double SF = 0.0, S1 = 0.0, S2 = 0.0;
    for (int i = 0; i < nF; ++i) SF += (double)pF[i];
    for (int i = 0; i < nd; ++i) { S1 += (double)pd1[i]; S2 += (double)pd2[i]; }
    const double off_diag = (SF - S2) / ((double)N * (double)(N - 1));
    const double diag     = 2.0 * S1 / (double)N;
    out[0] = (float)(off_diag - diag);
  }
}

extern "C" void kernel_launch(void* const* d_in, const int* in_sizes, int n_in,
                              void* d_out, int out_size, void* d_ws, size_t ws_size,
                              hipStream_t stream) {
  const float* x = (const float*)d_in[0];
  const float* y = (const float*)d_in[1];
  float* out = (float*)d_out;
  const int D = 768;
  const int N = in_sizes[0] / D;   // 8192

  float* ws  = (float*)d_ws;
  float* pF  = ws;         // 144 Frobenius partials
  float* pd1 = ws + 256;   // 64 row-dot sum partials
  float* pd2 = ws + 512;   // 64 row-dot sum-of-squares partials

  const int TB = D / TILE;               // 12
  dim3 grid(TB, TB);                     // 144 tiles of the 768x768 Gram pair
  gram_frob_kernel<<<grid, 128, 0, stream>>>(x, y, pF, N, D);
  rowdot_kernel<<<64, 256, 0, stream>>>(x, y, pd1, pd2, N, D);
  finalize_kernel<<<1, 32, 0, stream>>>(pF, TB * TB, pd1, pd2, 64, out, N);
}
